// EfficientAttention_34488587387511
// MI455X (gfx1250) — matrix operand order repaired and checked by reading, not compiled, not run on hardware
//
#include <hip/hip_runtime.h>

typedef _Float16 h16 __attribute__((ext_vector_type(16)));
typedef _Float16 h8  __attribute__((ext_vector_type(8)));
typedef _Float16 h4  __attribute__((ext_vector_type(4)));
typedef float    f8  __attribute__((ext_vector_type(8)));
typedef float    f4  __attribute__((ext_vector_type(4)));
typedef int      i4  __attribute__((ext_vector_type(4)));

#define NH 16
#define SEQ 4096
#define HD 64
#define KB 64    // keys per mainloop step
#define QT 2     // 16-query tiles per wave (32 queries/wave, 128/workgroup)

// LDS strides in halves; multiples of 8 halves (16B) for b128 alignment,
// odd multiple of 8 so 16-lane b128 patterns spread across banks.
constexpr int KS = 72;   // K tile row stride  (64 keys x 64 d, row-major)
constexpr int VS = 72;   // V^T tile row stride (64 d rows x 64 keys)

static __device__ __forceinline__ f8 wmma_f16(h16 a, h16 b, f8 c) {
    // (neg_a, A, neg_b, B, c_mod, C, reuse_a, reuse_b)
    return __builtin_amdgcn_wmma_f32_16x16x32_f16(false, a, false, b, (short)0, c,
                                                  false, false);
}

// Raw v_exp_f32: args are <= 0 here; underflow-to-zero is fine, skip libm fixup.
static __device__ __forceinline__ float fast_exp2(float x) {
#if __has_builtin(__builtin_amdgcn_exp2f)
    return __builtin_amdgcn_exp2f(x);
#else
    return exp2f(x);
#endif
}

// Swap values between the two 16-lane halves of a wave32 (lane ^ 16).
static __device__ __forceinline__ int xor16_i(int x) {
#if __has_builtin(__builtin_amdgcn_permlanex16)
    return __builtin_amdgcn_permlanex16(x, x, 0x76543210, 0xfedcba98, false, false);
#else
    return __shfl_xor(x, 16, 32);
#endif
}
static __device__ __forceinline__ float xor16_f(float x) {
    return __builtin_bit_cast(float, xor16_i(__builtin_bit_cast(int, x)));
}

static __device__ __forceinline__ f8 vmax8(f8 a, f8 b) {
#if __has_builtin(__builtin_elementwise_max)
    return __builtin_elementwise_max(a, b);
#else
    f8 r;
#pragma unroll
    for (int i = 0; i < 8; ++i) r[i] = fmaxf(a[i], b[i]);
    return r;
#endif
}
static __device__ __forceinline__ float hmax8(f8 v) {
    float a = fmaxf(fmaxf(v[0], v[1]), fmaxf(v[2], v[3]));
    float b = fmaxf(fmaxf(v[4], v[5]), fmaxf(v[6], v[7]));
    return fmaxf(a, b);
}
static __device__ __forceinline__ float hsum8(f8 v) {
    f4 t = {v[0] + v[4], v[1] + v[5], v[2] + v[6], v[3] + v[7]};
    return (t[0] + t[1]) + (t[2] + t[3]);
}

__global__ __launch_bounds__(128) void
EfficientAttention_34488587387511_kernel(const float* __restrict__ q,
                                         const float* __restrict__ k,
                                         const float* __restrict__ v,
                                         float* __restrict__ o) {
    __shared__ __align__(16) _Float16 K_lds[KB * KS];   // [key][d]
    __shared__ __align__(16) _Float16 V_lds[HD * VS];   // [d][key] (transposed)

    const int tid  = threadIdx.x;
    const int wave = tid >> 5;
    const int lane = tid & 31;
    const int l16  = lane & 15;
    const int hi   = lane >> 4;      // 0: lanes 0-15, 1: lanes 16-31

    const int nqblk = SEQ / (64 * QT);
    const int h     = blockIdx.x / nqblk;
    const int qb    = (blockIdx.x % nqblk) * (64 * QT);
    const int qr_w  = qb + wave * (16 * QT);   // this wave's 32 query rows

    const float* __restrict__ qh = q + (size_t)h * SEQ * HD;
    const float* __restrict__ kh = k + (size_t)h * SEQ * HD;
    const float* __restrict__ vh = v + (size_t)h * SEQ * HD;
    float*       __restrict__ oh = o + (size_t)h * SEQ * HD;

    // 1/sqrt(64) folded with log2(e) so softmax uses exp2 (v_exp_f32).
    const float qscale = 0.125f * 1.44269504088896340736f;

    // ---- Q^T B-fragments. B 32x16 in this kernel's k order: lane%16 = N(query), lane half
    // picks K(d) 0..15 / 16..31, contiguous in-lane.
    auto loadB_q = [&](int qt, int dbase) -> h16 {
        const float* p = qh + (size_t)(qr_w + qt * 16 + l16) * HD + dbase + hi * 16;
        h16 r;
#pragma unroll
        for (int g = 0; g < 4; ++g) {
            f4 x = *(const f4*)(p + 4 * g);
#pragma unroll
            for (int j = 0; j < 4; ++j) r[4 * g + j] = (_Float16)(x[j] * qscale);
        }
        return r;
    };
    h16 bq[QT][2];
#pragma unroll
    for (int qt = 0; qt < QT; ++qt) {
        bq[qt][0] = loadB_q(qt, 0);
        bq[qt][1] = loadB_q(qt, 32);
    }

    // A 16x32 f16 in this kernel's k order (B matches): lane%16 = M, lane half picks K 0..15 / 16..31,
    // contiguous in-lane: element j -> K = 16*hi + j.
    auto loadA = [&](const _Float16* base, int row, int stride, int koff) -> h16 {
        const _Float16* p = base + row * stride + koff + hi * 16;
        h8 lo = *(const h8*)(p);
        h8 hb = *(const h8*)(p + 8);
        return __builtin_shufflevector(lo, hb, 0, 1, 2, 3, 4, 5, 6, 7,
                                       8, 9, 10, 11, 12, 13, 14, 15);
    };

    // Build a P^T B-fragment (32 keys x 16 queries) from two exp'd score tiles:
    // own lane has keys {8hi+i} (ta) and {16+8hi+i} (tb); swap packed f16 pairs
    // with the partner half so each lane holds keys 16hi..16hi+15 contiguously.
    auto make_pb = [&](const f8& ta, const f8& tb) -> h16 {
        h8 hp0, hp1;
#pragma unroll
        for (int i = 0; i < 8; ++i) { hp0[i] = (_Float16)ta[i]; hp1[i] = (_Float16)tb[i]; }
        i4 i0 = __builtin_bit_cast(i4, hp0);
        i4 i1 = __builtin_bit_cast(i4, hp1);
        i4 x0, x1, blo, bhi;
#pragma unroll
        for (int j = 0; j < 4; ++j) { x0[j] = xor16_i(i0[j]); x1[j] = xor16_i(i1[j]); }
#pragma unroll
        for (int j = 0; j < 4; ++j) {
            blo[j] = hi ? x1[j] : i0[j];   // keys 16hi + 0..7
            bhi[j] = hi ? i1[j] : x0[j];   // keys 16hi + 8..15
        }
        h8 pblo = __builtin_bit_cast(h8, blo);
        h8 pbhi = __builtin_bit_cast(h8, bhi);
        return __builtin_shufflevector(pblo, pbhi, 0, 1, 2, 3, 4, 5, 6, 7,
                                       8, 9, 10, 11, 12, 13, 14, 15);
    };

    f8 acc[QT][4] = {};                     // O^T tiles: M=d (4x16), N=query
    float m_run[QT], l_run[QT];
#pragma unroll
    for (int qt = 0; qt < QT; ++qt) { m_run[qt] = -3.0e38f; l_run[qt] = 0.0f; }

    for (int kbase = 0; kbase < SEQ; kbase += KB) {
        __syncthreads();   // protect previous tile reads before overwrite
        // ---- Stage K row-major [key][d] (f32->f16).
#pragma unroll
        for (int it = 0; it < 8; ++it) {
            int idx = tid + it * 128;            // 1024 float4's
            int r = idx >> 4;                    // key 0..63
            int c = (idx & 15) * 4;              // d 0..60
            f4 kx = *(const f4*)(kh + (size_t)(kbase + r) * HD + c);
            h4 khv = {(_Float16)kx[0], (_Float16)kx[1], (_Float16)kx[2], (_Float16)kx[3]};
            *(h4*)&K_lds[r * KS + c] = khv;
        }
        // ---- Stage V transposed [d][key]: coalesced b32 loads, b64 LDS stores.
#pragma unroll
        for (int it = 0; it < 8; ++it) {
            int idx = tid + it * 128;            // 1024 half4-columns
            int c  = idx & 63;                   // d
            int r0 = (idx >> 6) * 4;             // key 0,4,..,60
            h4 vv;
#pragma unroll
            for (int j = 0; j < 4; ++j)
                vv[j] = (_Float16)vh[(size_t)(kbase + r0 + j) * HD + c];
            *(h4*)&V_lds[c * VS + r0] = vv;
        }
        __syncthreads();

        // ---- S^T = K * Q^T : 4 key groups x QT query tiles; each K A-frag
        // pair feeds QT WMMAs.
        f8 sc[QT][4];
#pragma unroll
        for (int t = 0; t < 4; ++t) {
            h16 a0 = loadA(K_lds, t * 16 + l16, KS, 0);
            h16 a1 = loadA(K_lds, t * 16 + l16, KS, 32);
#pragma unroll
            for (int qt = 0; qt < QT; ++qt) {
                f8 z = {};
                z = wmma_f16(a0, bq[qt][0], z);
                sc[qt][t] = wmma_f16(a1, bq[qt][1], z);
            }
        }
        // Lane holds, for query l16 (tile qt): keys {16t + 8hi + i} in sc[qt][t][i].

        // ---- Online softmax per query tile: in-lane reduce + one half-swap.
        h16 pb[QT][2];
#pragma unroll
        for (int qt = 0; qt < QT; ++qt) {
            f8 mv = vmax8(vmax8(sc[qt][0], sc[qt][1]), vmax8(sc[qt][2], sc[qt][3]));
            float mx = hmax8(mv);
            mx = fmaxf(mx, xor16_f(mx));
            float mnew  = fmaxf(m_run[qt], mx);
            float alpha = fast_exp2(m_run[qt] - mnew);
            m_run[qt] = mnew;

            f8 pe[4];
#pragma unroll
            for (int t = 0; t < 4; ++t) {
                f8 e = sc[qt][t] - mnew;          // vector - scalar -> v_pk_add
#pragma unroll
                for (int i = 0; i < 8; ++i) pe[t][i] = fast_exp2(e[i]);
            }
            f8 sv = (pe[0] + pe[1]) + (pe[2] + pe[3]);
            float s = hsum8(sv);
            s += xor16_f(s);
            l_run[qt] = l_run[qt] * alpha + s;

            pb[qt][0] = make_pb(pe[0], pe[1]);    // keys 0-31
            pb[qt][1] = make_pb(pe[2], pe[3]);    // keys 32-63

            // Rescale running O^T (alpha is a per-lane scalar: N=query=l16).
#pragma unroll
            for (int f = 0; f < 4; ++f) acc[qt][f] = acc[qt][f] * alpha;
        }

        // ---- O^T += V^T * P^T : each V A-frag feeds QT WMMAs.
#pragma unroll
        for (int f = 0; f < 4; ++f) {
            h16 av0 = loadA(V_lds, f * 16 + l16, VS, 0);
#pragma unroll
            for (int qt = 0; qt < QT; ++qt) acc[qt][f] = wmma_f16(av0, pb[qt][0], acc[qt][f]);
            h16 av1 = loadA(V_lds, f * 16 + l16, VS, 32);
#pragma unroll
            for (int qt = 0; qt < QT; ++qt) acc[qt][f] = wmma_f16(av1, pb[qt][1], acc[qt][f]);
        }
    }

    // ---- Normalize (per-lane scalar) and store; d runs are contiguous per lane.
#pragma unroll
    for (int qt = 0; qt < QT; ++qt) {
        float inv = 1.0f / l_run[qt];
        float* orow = oh + (size_t)(qr_w + qt * 16 + l16) * HD;
#pragma unroll
        for (int f = 0; f < 4; ++f) {
            int d0 = f * 16 + 8 * hi;           // acc[qt][f][i] -> d = d0 + i
            f4 o0, o1;
#pragma unroll
            for (int i = 0; i < 4; ++i) {
                o0[i] = acc[qt][f][i] * inv;
                o1[i] = acc[qt][f][i + 4] * inv;
            }
            *(f4*)(orow + d0)     = o0;
            *(f4*)(orow + d0 + 4) = o1;
        }
    }
}

extern "C" void kernel_launch(void* const* d_in, const int* in_sizes, int n_in,
                              void* d_out, int out_size, void* d_ws, size_t ws_size,
                              hipStream_t stream) {
    const float* q = (const float*)d_in[0];
    const float* k = (const float*)d_in[1];
    const float* v = (const float*)d_in[2];
    float* o = (float*)d_out;
    dim3 grid(NH * (SEQ / (64 * QT)));
    EfficientAttention_34488587387511_kernel<<<grid, 128, 0, stream>>>(q, k, v, o);
}
